// GraphConv_28475633173124
// MI455X (gfx1250) — matrix operand order repaired and checked by reading, not compiled, not run on hardware
//
#include <hip/hip_runtime.h>

typedef __attribute__((ext_vector_type(16))) _Float16 v16h;
typedef __attribute__((ext_vector_type(8)))  _Float16 v8h;
typedef __attribute__((ext_vector_type(8)))  float    v8f;

// Problem shape (fixed by setup_inputs)
constexpr int B     = 4;
constexpr int C     = 128;
constexpr int N     = 4096;   // 2^12
constexpr int K     = 16;
constexpr int CO    = 256;
constexpr int TWO_C = 256;    // 2*C, GEMM reduction dim
constexpr int NSLOT = 64;     // contention-splitting slots for BN stat atomics
constexpr int HPAD  = TWO_C + 8;  // LDS row stride (f16): 528B/row -> 4-bank shift, conflict-free
constexpr int NPB   = 8;      // nodes per block (weight-register reuse factor)
constexpr float EPS = 1e-5f;

// ws layout (bytes)
constexpr size_t OFF_W     = 0;          // wf16   [CO][2C] f16   : 128 KB
constexpr size_t OFF_SUMS  = 131072;     // sums   [NSLOT][CO]    :  64 KB
constexpr size_t OFF_SUMSQ = 196608;     // sumsq  [NSLOT][CO]    :  64 KB
constexpr size_t OFF_STATS = 262144;     // stats  [2][CO]        :   2 KB
constexpr size_t OFF_XT    = 294912;     // xT     [B][N][C] f16  :   4 MB
constexpr size_t XT_BYTES  = (size_t)B * N * C * 2;

// ---------------------------------------------------------------------------
// Kernel 0: convert W [CO][2C] f32 -> f16 in ws, zero the stat accumulators.
// ---------------------------------------------------------------------------
__global__ void prep_kernel(const float* __restrict__ w,
                            _Float16* __restrict__ wf16,
                            float* __restrict__ sums,
                            float* __restrict__ sumsq) {
    int tid = blockIdx.x * blockDim.x + threadIdx.x;     // 0..65535
    wf16[tid] = (_Float16)w[tid];
    if (tid < NSLOT * CO) { sums[tid] = 0.0f; sumsq[tid] = 0.0f; }
}

// ---------------------------------------------------------------------------
// Kernel 0b: transpose x [B,C,N] f32 -> xT [B,N,C] f16 (coalesced reads,
// 64B-contiguous writes per thread). Makes the main-kernel gather vectorizable.
// ---------------------------------------------------------------------------
__global__ void transpose_kernel(const float* __restrict__ x,
                                 _Float16* __restrict__ xT) {
    const int p   = blockIdx.x * 256 + threadIdx.x;      // b*N + n
    const int b   = p >> 12;
    const int cb0 = blockIdx.y * 32;                     // channel slab
    const float* src = x + (size_t)b * C * N + (p & (N - 1));
    _Float16*    dst = xT + (size_t)p * C;
#pragma unroll
    for (int cb = 0; cb < 32; cb += 8) {
        v8h v;
#pragma unroll
        for (int j = 0; j < 8; ++j)
            v[j] = (_Float16)src[(size_t)(cb0 + cb + j) * N];
        *(v8h*)(dst + cb0 + cb) = v;
    }
}

// ---------------------------------------------------------------------------
// Kernel 1: one workgroup per 8 nodes, 8 waves. Register-resident weights
// (16 x v16h / wave). Pipeline per node t:
//   [issue gather for t+1 into regs]        -> LOADcnt only
//   [ds_load A frags + 16 WMMAs for t]      -> DScnt partial waits only
//   [pack + ds_store gather of t+1 to LDS]  -> waits gather LOADcnt here
//   [reduce, store, atomics]  [barrier]
// TR=true: gather reads f16 node-major xT (2x global_load_b128 / thread);
// TR=false: fallback strided f32 gather straight from x.
// ---------------------------------------------------------------------------
template <bool TR>
__global__ void __launch_bounds__(256) edgeconv_kernel(
        const float*     __restrict__ x,      // [B, C, N]
        const _Float16*  __restrict__ xT,     // [B, N, C] (TR only)
        const long long* __restrict__ eidx,   // [2, B, N, K] int64
        const _Float16*  __restrict__ wf16,   // [CO, 2C] f16
        const float*     __restrict__ bias,   // [CO]
        float*           __restrict__ out_raw,// [B, CO, N] (pre-BN max)
        float*           __restrict__ sums,   // [NSLOT, CO]
        float*           __restrict__ sumsq)  // [NSLOT, CO]
{
    __shared__ _Float16 hs[2][K * HPAD];      // double buffer, 2 x 8448 B

    const int lane = threadIdx.x & 31;
    const int wv   = threadIdx.x >> 5;        // 0..7 : wave -> channel group
    const int col  = lane & 15;
    const int hi   = lane >> 4;               // half-wave select
    const int o0   = wv * 32 + col;
    const int o1   = o0 + 16;

    // ---- Register-resident B fragments (lane = column, K run of 16/half) ----
    v16h bf0[8], bf1[8];
#pragma unroll
    for (int kk = 0; kk < 8; ++kk) {
        const int kb = kk * 32 + hi * 16;
        bf0[kk] = *(const v16h*)(wf16 + (size_t)o0 * TWO_C + kb);
        bf1[kk] = *(const v16h*)(wf16 + (size_t)o1 * TWO_C + kb);
    }
    const float bias0 = bias[o0];
    const float bias1 = bias[o1];

    // gather-stage mapping: thread -> (row gk, contiguous channel chunk c0..c0+7)
    const int gk    = threadIdx.x >> 4;       // 0..15 : neighbor row
    const int c0    = (threadIdx.x & 15) * 8; // contiguous 8-channel chunk
    const int rowsw = gk * HPAD;              // LDS store row offset
    const int rowld = (lane & 15) * HPAD;     // LDS load row offset (A frags)
    const int p0    = blockIdx.x * NPB;

    float gxf[16];                            // in-flight gather (fallback path)
    v8h   gxi = {}, gxj = {};                 // in-flight gather (f16 path)

    // issue gather loads for node p0+t (registers only, no LDS traffic)
    auto gather = [&](int t) {
        const int p = p0 + t;
        const int b = p >> 12;
        const int n = p & (N - 1);
        const long long base = ((long long)b * N + n) * K + gk;
        const int nj = (int)eidx[base];                             // edge_index[0]
        const int ni = (int)eidx[(long long)B * N * K + base];      // edge_index[1]
        if constexpr (TR) {
            const _Float16* xb = xT + (size_t)b * N * C + c0;
            gxi = *(const v8h*)(xb + (size_t)ni * C);               // x_i chunk
            gxj = *(const v8h*)(xb + (size_t)nj * C);               // x_j chunk
        } else {
            const float* xb = x + (size_t)b * C * N + (size_t)c0 * N;
#pragma unroll
            for (int j = 0; j < 8; ++j) {
                gxf[j]     = xb[(size_t)j * N + ni];
                gxf[8 + j] = xb[(size_t)j * N + nj];
            }
        }
    };

    // pack gathered regs and store into LDS buffer `buf`
    auto stash = [&](int buf) {
        v8h hx, hd;
        if constexpr (TR) {
            hx = gxi;
            hd = gxj - gxi;                   // packed f16 subtract
        } else {
#pragma unroll
            for (int j = 0; j < 8; ++j) {
                hx[j] = (_Float16)gxf[j];
                hd[j] = (_Float16)(gxf[8 + j] - gxf[j]);
            }
        }
        if (buf == 0) {
            *(v8h*)&hs[0][rowsw + c0]       = hx;
            *(v8h*)&hs[0][rowsw + 128 + c0] = hd;
        } else {
            *(v8h*)&hs[1][rowsw + c0]       = hx;
            *(v8h*)&hs[1][rowsw + 128 + c0] = hd;
        }
    };

    gather(0);
    stash(0);
    __syncthreads();

#pragma unroll
    for (int t = 0; t < NPB; ++t) {
        const int buf = t & 1;                // literal after unroll -> pure DS ops
        if (t + 1 < NPB) gather(t + 1);       // LOADcnt-only prefetch

        // ---- 16 WMMAs for node t (A from LDS, B from registers) ----
        v8f acc0 = {}; v8f acc1 = {};
#pragma unroll
        for (int kk = 0; kk < 8; ++kk) {
            // A fragment (16-bit A 16x32): K contiguous run hi*16..hi*16+15 (matches B)
            const int ka = kk * 32 + hi * 16;
            v8h a0 = *(const v8h*)&hs[buf][rowld + ka];
            v8h a1 = *(const v8h*)&hs[buf][rowld + ka + 8];
            v16h a = __builtin_shufflevector(a0, a1,
                     0,1,2,3,4,5,6,7,8,9,10,11,12,13,14,15);
            acc0 = __builtin_amdgcn_wmma_f32_16x16x32_f16(
                       false, a, false, bf0[kk], (short)0, acc0, false, false);
            acc1 = __builtin_amdgcn_wmma_f32_16x16x32_f16(
                       false, a, false, bf1[kk], (short)0, acc1, false, false);
        }

        // ---- stash node t+1 gather into the other buffer ----
        if (t + 1 < NPB) stash(buf ^ 1);

        // ---- bias + relu, reduce over the 16 neighbor rows ----
        float m0 = 0.0f, m1 = 0.0f;           // relu >= 0, so 0 is a safe init
        float s0 = 0.0f, s1 = 0.0f, q0 = 0.0f, q1 = 0.0f;
#pragma unroll
        for (int r = 0; r < 8; ++r) {         // rows r (lanes 0-15) / r+8 (16-31)
            const float v0 = fmaxf(acc0[r] + bias0, 0.0f);
            const float v1 = fmaxf(acc1[r] + bias1, 0.0f);
            m0 = fmaxf(m0, v0);  m1 = fmaxf(m1, v1);
            s0 += v0;            s1 += v1;
            q0 += v0 * v0;       q1 += v1 * v1;
        }
        // lanes L and L+16 hold the same channel (rows 0-7 vs 8-15)
        m0 = fmaxf(m0, __shfl_xor(m0, 16, 32));
        m1 = fmaxf(m1, __shfl_xor(m1, 16, 32));
        s0 += __shfl_xor(s0, 16, 32);  s1 += __shfl_xor(s1, 16, 32);
        q0 += __shfl_xor(q0, 16, 32);  q1 += __shfl_xor(q1, 16, 32);

        if (hi == 0) {
            const int p = p0 + t;
            const int b = p >> 12;
            const int n = p & (N - 1);
            const int slot = p & (NSLOT - 1);
            out_raw[((size_t)b * CO + o0) * N + n] = m0;
            out_raw[((size_t)b * CO + o1) * N + n] = m1;
            atomicAdd(&sums [slot * CO + o0], s0);
            atomicAdd(&sums [slot * CO + o1], s1);
            atomicAdd(&sumsq[slot * CO + o0], q0);
            atomicAdd(&sumsq[slot * CO + o1], q1);
        }
        __syncthreads();
    }
}

// ---------------------------------------------------------------------------
// Kernel 2: fold slots -> per-channel mean & (gamma * rsqrt(var+eps)).
// ---------------------------------------------------------------------------
__global__ void stats_kernel(const float* __restrict__ sums,
                             const float* __restrict__ sumsq,
                             const float* __restrict__ gamma,
                             float* __restrict__ stats) {  // [2][CO]
    const int o = threadIdx.x;
    float S = 0.0f, Q = 0.0f;
    for (int s = 0; s < NSLOT; ++s) {
        S += sums [s * CO + o];
        Q += sumsq[s * CO + o];
    }
    const float cnt  = (float)B * (float)N * (float)K;
    const float mean = S / cnt;
    const float var  = Q / cnt - mean * mean;
    stats[o]      = mean;
    stats[CO + o] = gamma[o] * rsqrtf(var + EPS);
}

// ---------------------------------------------------------------------------
// Kernel 3: in-place affine on the K-max (valid since BN scale > 0).
// ---------------------------------------------------------------------------
__global__ void finalize_kernel(float* __restrict__ out,
                                const float* __restrict__ stats,
                                const float* __restrict__ beta) {
    const int i = blockIdx.x * blockDim.x + threadIdx.x;  // < B*CO*N
    const int o = (i >> 12) & (CO - 1);                   // N == 2^12
    out[i] = (out[i] - stats[o]) * stats[CO + o] + beta[o];
}

// ---------------------------------------------------------------------------
extern "C" void kernel_launch(void* const* d_in, const int* in_sizes, int n_in,
                              void* d_out, int out_size, void* d_ws, size_t ws_size,
                              hipStream_t stream) {
    const float*     x     = (const float*)d_in[0];     // [B,C,N,1]
    const float*     w     = (const float*)d_in[1];     // [CO,2C]
    const float*     bias  = (const float*)d_in[2];     // [CO]
    const float*     gamma = (const float*)d_in[3];     // [CO]
    const float*     beta  = (const float*)d_in[4];     // [CO]
    const long long* eidx  = (const long long*)d_in[5]; // [2,B,N,K] int64

    char* ws = (char*)d_ws;
    _Float16* wf16  = (_Float16*)(ws + OFF_W);
    float*    sums  = (float*)  (ws + OFF_SUMS);
    float*    sumsq = (float*)  (ws + OFF_SUMSQ);
    float*    stats = (float*)  (ws + OFF_STATS);
    _Float16* xT    = (_Float16*)(ws + OFF_XT);
    float*    out   = (float*)d_out;                    // [B,CO,N]

    const bool tr = ws_size >= OFF_XT + XT_BYTES;

    prep_kernel<<<(CO * TWO_C) / 256, 256, 0, stream>>>(w, wf16, sums, sumsq);
    if (tr) {
        transpose_kernel<<<dim3((B * N) / 256, C / 32), 256, 0, stream>>>(x, xT);
        edgeconv_kernel<true><<<(B * N) / NPB, 256, 0, stream>>>(
            x, xT, eidx, wf16, bias, out, sums, sumsq);
    } else {
        edgeconv_kernel<false><<<(B * N) / NPB, 256, 0, stream>>>(
            x, xT, eidx, wf16, bias, out, sums, sumsq);
    }
    stats_kernel   <<<1,                  CO,  0, stream>>>(sums, sumsq, gamma, stats);
    finalize_kernel<<<(B * CO * N) / 256, 256, 0, stream>>>(out, stats, beta);
}
